// Conv_Layer_35905926594765
// MI455X (gfx1250) — compile-verified, matrix operand order checked by reading, not run on hardware
//
#include <hip/hip_runtime.h>
#include <hip/hip_bf16.h>

typedef __attribute__((ext_vector_type(16))) _Float16 v16h;
typedef __attribute__((ext_vector_type(8)))  _Float16 v8h;
typedef __attribute__((ext_vector_type(4)))  _Float16 v4h;
typedef __attribute__((ext_vector_type(8)))  float    v8f;

#define BATCH 32
#define SEQ   1024
#define CH    256
#define SP    (SEQ + 16)     // haloed row: 8 zeros each side
#define EPSN  1e-5f
#define LA    0.1f
#define NT    64             // position-tile per workgroup

// ---------------------------------------------------------------------------
// Zero the 8-element halos of the f16 activation buffer (once per launch).
// ---------------------------------------------------------------------------
__global__ void zero_halo_kernel(_Float16* __restrict__ y) {
    int idx = blockIdx.x * 256 + threadIdx.x;
    if (idx >= BATCH * CH * 16) return;
    int row = idx >> 4, j = idx & 15;
    int off = (j < 8) ? j : (SEQ + j);          // [0,8) and [S+8, S+16)
    y[(size_t)row * SP + off] = (_Float16)0.f;
}

// ---------------------------------------------------------------------------
// Convert conv weight [COUT][CIN][KW] f32 -> f16 permuted t-major:
// wbuf[cout][t][c]  so that the GEMM K-index k = t*CH + c is contiguous.
// ---------------------------------------------------------------------------
template<int KW>
__global__ void wcvt_kernel(const float* __restrict__ w, _Float16* __restrict__ o) {
    int i = blockIdx.x * 256 + threadIdx.x;
    if (i >= CH * CH * KW) return;
    int c    = i & (CH - 1);
    int rest = i >> 8;            // cout*KW + t
    int t    = rest % KW;
    int cout = rest / KW;
    o[i] = (_Float16)w[(cout * CH + c) * KW + t];
}

// ---------------------------------------------------------------------------
// ChannelNorm (unbiased var over C) + ReLU + cvt to f16 into haloed buffer.
// FIRST=true reads the original [B,S,C] input; else reads [B,C,S] fp32.
// ---------------------------------------------------------------------------
template<bool FIRST>
__global__ __launch_bounds__(256)
void cnorm_relu_kernel(const float* __restrict__ src,
                       const float* __restrict__ cw,
                       const float* __restrict__ cb,
                       _Float16* __restrict__ y) {
    int p = blockIdx.x * 256 + threadIdx.x;     // p = b*S + l
    if (p >= BATCH * SEQ) return;
    int b = p >> 10, l = p & (SEQ - 1);
    float sum = 0.f, sq = 0.f;
    for (int c = 0; c < CH; ++c) {
        float v = FIRST ? src[(size_t)p * CH + c]
                        : src[(((size_t)b * CH + c) << 10) + l];
        sum += v; sq += v * v;
    }
    float m   = sum * (1.f / CH);
    float var = (sq - CH * m * m) * (1.f / (CH - 1));   // ddof = 1
    float r   = rsqrtf(var + EPSN);
    for (int c = 0; c < CH; ++c) {
        float v = FIRST ? src[(size_t)p * CH + c]
                        : src[(((size_t)b * CH + c) << 10) + l];
        float t = (v - m) * r * cw[c] + cb[c];
        t = fmaxf(t, 0.f);
        y[((size_t)b * CH + c) * SP + 8 + l] = (_Float16)t;
    }
}

// ---------------------------------------------------------------------------
// Implicit-GEMM conv + fused ResidualAttention gate + pad_to.
// Grid: BATCH*(SEQ/NT) workgroups of 256 threads (8 waves).
// Workgroup computes D[256 x 64]; each wave a 32x64 sub-tile (8 WMMA accums).
// K-order is t-major (k = t*256 + c): each 32-chunk has constant t and
// contiguous c -> A and B fragments are aligned 16B vector loads.
// ---------------------------------------------------------------------------
template<int KW, int PAD>
__global__ __launch_bounds__(256)
void conv_gate_kernel(const _Float16* __restrict__ wf,   // [CH][KW][CH] f16
                      const _Float16* __restrict__ y,    // [B][CH][SP] f16
                      const float*    __restrict__ fc,   // [2][CH] f32
                      float*          __restrict__ out)  // [B][CH][SEQ] f32
{
    constexpr int KTOT = CH * KW;
    constexpr int LOUT = SEQ + 2 * PAD - KW + 1;   // pad_to zeroes l >= LOUT
    constexpr int WIN  = NT + KW - 1;              // input window per tile
    constexpr int WINQ = (WIN + 3) / 4;            // window in 4-half quads

    const int wg   = blockIdx.x;
    const int b    = wg >> 4;                 // SEQ/NT = 16 tiles per batch
    const int n0   = (wg & 15) << 6;
    const int wave = threadIdx.x >> 5;
    const int lane = threadIdx.x & 31;
    const int lrow = lane & 15;
    const int kb   = (lane < 16) ? 0 : 8;     // K-half held by this lane
    const int mtop = (lane >> 4) << 3;        // C/D layout: +8 rows for hi lanes
    const int mbase = wave << 5;              // 32 output rows per wave

    // Stage activation window [WIN positions][256 ch] into LDS, pos-major,
    // +8 half pad per row to spread banks. 8B global loads, 2B LDS scatter.
    __shared__ _Float16 ytile[72][CH + 8];
    const _Float16* ybase = y + (size_t)b * CH * SP + 8;
    for (int idx = threadIdx.x; idx < CH * WINQ; idx += 256) {
        int c  = idx / WINQ;
        int jq = idx - c * WINQ;
        v4h q = *(const v4h*)(ybase + (size_t)c * SP + (n0 - PAD) + jq * 4);
#pragma unroll
        for (int u = 0; u < 4; ++u) ytile[jq * 4 + u][c] = q[u];
    }
    __syncthreads();

    v8f acc[2][4] = {};

    for (int k0 = 0; k0 < KTOT; k0 += 32) {
        const int t     = k0 >> 8;       // constant within the chunk
        const int cbase = k0 & 255;

        // A fragments: two aligned 16B global loads per 16x32 tile
        v16h a[2];
#pragma unroll
        for (int mi = 0; mi < 2; ++mi) {
            const _Float16* ap = wf + (size_t)(mbase + mi * 16 + lrow) * KTOT + k0 + kb;
            v8h lo = *(const v8h*)ap;          // K = kb+0..7
            v8h hi = *(const v8h*)(ap + 16);   // K = kb+16..23
            v16h av;
#pragma unroll
            for (int i = 0; i < 8; ++i) { av[i] = lo[i]; av[i + 8] = hi[i]; }
            a[mi] = av;
        }

        // B fragments from LDS, one sub-tile at a time (two 16B ds loads)
#pragma unroll
        for (int ni = 0; ni < 4; ++ni) {
            int j = ni * 16 + lrow + t;        // position within window
            const _Float16* bp = &ytile[j][cbase + kb];
            v8h lo = *(const v8h*)bp;          // c = cbase+kb+0..7
            v8h hi = *(const v8h*)(bp + 16);   // c = cbase+kb+16..23
            v16h bv;
#pragma unroll
            for (int i = 0; i < 8; ++i) { bv[i] = lo[i]; bv[i + 8] = hi[i]; }
#pragma unroll
            for (int mi = 0; mi < 2; ++mi)
                acc[mi][ni] = __builtin_amdgcn_wmma_f32_16x16x32_f16(
                    false, a[mi], false, bv, (short)0, acc[mi][ni],
                    false, false);
        }
    }

    // ---- Fused gate: y_cls[n] = sum_M fc[cls][M] * h[M][n] over all 256 M ----
    __shared__ float red[8][4][16][2];
    __shared__ float scl[4][16];
#pragma unroll
    for (int ni = 0; ni < 4; ++ni) {
        float p0 = 0.f, p1 = 0.f;
#pragma unroll
        for (int mi = 0; mi < 2; ++mi)
#pragma unroll
            for (int r = 0; r < 8; ++r) {
                int M   = mbase + mi * 16 + mtop + r;
                float h = acc[mi][ni][r];
                p0 += fc[M]      * h;
                p1 += fc[CH + M] * h;
            }
        // lanes L and L+16 hold the same column N=L&15 -> fold
        p0 += __shfl_xor(p0, 16);
        p1 += __shfl_xor(p1, 16);
        if (lane < 16) { red[wave][ni][lane][0] = p0; red[wave][ni][lane][1] = p1; }
    }
    __syncthreads();
    if (threadIdx.x < 64) {
        int ni = threadIdx.x >> 4, n = threadIdx.x & 15;
        float y0 = 0.f, y1 = 0.f;
#pragma unroll
        for (int w = 0; w < 8; ++w) { y0 += red[w][ni][n][0]; y1 += red[w][ni][n][1]; }
        float s   = 0.5f * (y0 + y1) + LA * fmaxf(y0, y1);
        float sig = 1.f / (1.f + __expf(-s));
        int ng = n0 + ni * 16 + n;
        scl[ni][n] = (ng < LOUT) ? (1.f + sig) : 0.f;   // gate + pad_to zeroing
    }
    __syncthreads();

    // ---- Store h * (1 + score) as fp32 [B][CH][SEQ] ----
    float* obase = out + (size_t)b * CH * SEQ;
#pragma unroll
    for (int mi = 0; mi < 2; ++mi)
#pragma unroll
        for (int ni = 0; ni < 4; ++ni) {
            float s  = scl[ni][lrow];
            int   ng = n0 + ni * 16 + lrow;
#pragma unroll
            for (int r = 0; r < 8; ++r) {
                int M = mbase + mi * 16 + mtop + r;
                obase[(size_t)M * SEQ + ng] = acc[mi][ni][r] * s;
            }
        }
}

// ---------------------------------------------------------------------------
// Final transpose [B][C][S] -> [B][S][C] (fp32), coalesced writes.
// ---------------------------------------------------------------------------
__global__ void transpose_out_kernel(const float* __restrict__ x, float* __restrict__ o) {
    size_t idx = (size_t)blockIdx.x * 256 + threadIdx.x;
    if (idx >= (size_t)BATCH * SEQ * CH) return;
    int c    = (int)(idx & 255);
    size_t r = idx >> 8;
    int l    = (int)(r & (SEQ - 1));
    int b    = (int)(r >> 10);
    o[idx] = x[(((size_t)b << 8) + c) * SEQ + l];
}

// ---------------------------------------------------------------------------
template<int KW, int PAD>
static inline void run_block(void* const* d_in, int wi, int ci, int bi,
                             const float* src, bool first, const float* fc,
                             float* xbuf, _Float16* ybuf, _Float16* wbuf,
                             hipStream_t stream) {
    const float* W  = (const float*)d_in[wi];
    const float* cw = (const float*)d_in[ci];
    const float* cb = (const float*)d_in[bi];
    int wn = CH * CH * KW;
    wcvt_kernel<KW><<<(wn + 255) / 256, 256, 0, stream>>>(W, wbuf);
    if (first)
        cnorm_relu_kernel<true ><<<(BATCH * SEQ) / 256, 256, 0, stream>>>(src, cw, cb, ybuf);
    else
        cnorm_relu_kernel<false><<<(BATCH * SEQ) / 256, 256, 0, stream>>>(src, cw, cb, ybuf);
    conv_gate_kernel<KW, PAD><<<BATCH * (SEQ / NT), 256, 0, stream>>>(wbuf, ybuf, fc, xbuf);
}

extern "C" void kernel_launch(void* const* d_in, const int* in_sizes, int n_in,
                              void* d_out, int out_size, void* d_ws, size_t ws_size,
                              hipStream_t stream) {
    (void)in_sizes; (void)n_in; (void)out_size; (void)ws_size;
    const float* inp = (const float*)d_in[0];
    const float* fc  = (const float*)d_in[16];

    char* ws = (char*)d_ws;
    const size_t xbytes = (size_t)BATCH * CH * SEQ * 4;   // 33.5 MB fp32 ping
    const size_t ybytes = (size_t)BATCH * CH * SP  * 2;   // 16.3 MB f16 haloed
    float*    xbuf = (float*)ws;
    _Float16* ybuf = (_Float16*)(ws + xbytes);
    _Float16* wbuf = (_Float16*)(ws + xbytes + ybytes);   // 1 MB f16 weights

    int halo = BATCH * CH * 16;
    zero_halo_kernel<<<(halo + 255) / 256, 256, 0, stream>>>(ybuf);

    run_block<4, 1>(d_in,  3,  1,  2, inp,  true,  fc, xbuf, ybuf, wbuf, stream);
    run_block<5, 2>(d_in,  6,  4,  5, xbuf, false, fc, xbuf, ybuf, wbuf, stream);
    run_block<6, 2>(d_in,  9,  7,  8, xbuf, false, fc, xbuf, ybuf, wbuf, stream);
    run_block<7, 3>(d_in, 12, 10, 11, xbuf, false, fc, xbuf, ybuf, wbuf, stream);
    run_block<8, 3>(d_in, 15, 13, 14, xbuf, false, fc, xbuf, ybuf, wbuf, stream);

    size_t total = (size_t)BATCH * SEQ * CH;
    transpose_out_kernel<<<(unsigned)((total + 255) / 256), 256, 0, stream>>>(xbuf, (float*)d_out);
}
